// GraphConstructor_46514495816145
// MI455X (gfx1250) — matrix operand order repaired and checked by reading, not compiled, not run on hardware
//
#include <hip/hip_runtime.h>

typedef __attribute__((ext_vector_type(8)))  _Float16 v8h;
typedef __attribute__((ext_vector_type(16))) _Float16 v16h;
typedef __attribute__((ext_vector_type(8)))  float    v8f;

#define N_TOTAL 8192
#define DIM     64
#define KNB     5
#define THRESH  0.1f
#define SIMW    36           // padded LDS row stride (dwords)

// ---- output layout (4-byte elements, concatenated in return order) ----
#define OFF_SRC_LOOP 40960
#define OFF_DST      49152
#define OFF_DST_LOOP 90112
#define OFF_W        98304
#define OFF_W_LOOP   139264

// Single-wave workgroup: DS ops are in-order within a wave (ISA 15.5), so
// cross-lane LDS store->load only needs a *compiler* ordering fence, not a
// barrier (which would also force s_wait_loadcnt 0 and kill the B prefetch).
__device__ __forceinline__ void wave_lds_fence() { asm volatile("" ::: "memory"); }

// ------------------------------------------------------------------
// Kernel 1: row-normalize (f32) and convert to f16 feature matrix.
// ------------------------------------------------------------------
__global__ void __launch_bounds__(256)
normalize_f16_kernel(const float* __restrict__ x, _Float16* __restrict__ xh) {
    int row = blockIdx.x * blockDim.x + threadIdx.x;
    if (row >= N_TOTAL) return;
    const float4* xp = (const float4*)(x + (size_t)row * DIM);
    float4 v[16];
    float s = 0.0f;
#pragma unroll
    for (int i = 0; i < 16; ++i) {
        v[i] = xp[i];
        s += v[i].x * v[i].x + v[i].y * v[i].y + v[i].z * v[i].z + v[i].w * v[i].w;
    }
    float inv = 1.0f / sqrtf(s + 1e-20f);
    v8h* o = (v8h*)(xh + (size_t)row * DIM);
#pragma unroll
    for (int i = 0; i < 8; ++i) {
        float4 a = v[2 * i], b = v[2 * i + 1];
        v8h h;
        h[0] = (_Float16)(a.x * inv); h[1] = (_Float16)(a.y * inv);
        h[2] = (_Float16)(a.z * inv); h[3] = (_Float16)(a.w * inv);
        h[4] = (_Float16)(b.x * inv); h[5] = (_Float16)(b.y * inv);
        h[6] = (_Float16)(b.z * inv); h[7] = (_Float16)(b.w * inv);
        o[i] = h;
    }
}

__device__ __forceinline__ void topk_insert(float v, int idx,
                                            float (&tv)[KNB], int (&ti)[KNB]) {
    if (v > tv[KNB - 1]) {
        tv[KNB - 1] = v; ti[KNB - 1] = idx;
#pragma unroll
        for (int t = KNB - 2; t >= 0; --t) {
            if (tv[t + 1] > tv[t]) {
                float fv = tv[t]; tv[t] = tv[t + 1]; tv[t + 1] = fv;
                int   fi = ti[t]; ti[t] = ti[t + 1]; ti[t + 1] = fi;
            }
        }
    }
}

// ------------------------------------------------------------------
// Kernel 2: fused Gram tile (WMMA f16->f32) + thresholded top-5.
// One wave32 per 16-row tile; 2 column tiles (32 cols) per iteration.
// ------------------------------------------------------------------
__global__ void __launch_bounds__(32)
topk_graph_kernel(const _Float16* __restrict__ xh,
                  int* __restrict__ iout, float* __restrict__ fout) {
    __shared__ float sim[16 * SIMW];   // 16 rows x 32 cols (+pad), stride 36
    __shared__ float mval[32][KNB];
    __shared__ int   midx[32][KNB];

    const int lane = threadIdx.x;      // 0..31
    const int m    = lane & 15;
    const int hi   = lane >> 4;
    const int rowBase = blockIdx.x * 16;
    const int grow = rowBase + m;
    const int diagJJ = rowBase >> 5;   // iteration containing the self column

    // ---- A tile resident in VGPRs (contiguous k = 16*hi + s, matching B) ----
    const _Float16* arow = xh + (size_t)grow * DIM;
    v16h a0, a1;
    {
        v8h l0 = *(const v8h*)(arow + hi * 16);
        v8h h0 = *(const v8h*)(arow + 8 + hi * 16);
        v8h l1 = *(const v8h*)(arow + 32 + hi * 16);
        v8h h1 = *(const v8h*)(arow + 40 + hi * 16);
#pragma unroll
        for (int i = 0; i < 8; ++i) {
            a0[i] = l0[i]; a0[i + 8] = h0[i];
            a1[i] = l1[i]; a1[i + 8] = h1[i];
        }
    }

    // Sentinel = THRESH: reference discards entries <= THRESH anyway, so
    // top-5 of {v > THRESH} is equivalent after thresholding.
    float tv[KNB]; int ti[KNB];
#pragma unroll
    for (int t = 0; t < KNB; ++t) { tv[t] = THRESH; ti[t] = -1; }

    // ---- prefetch B tiles for jj = 0 ----
    const _Float16* bbase = xh + (size_t)m * DIM + hi * 16;
    v16h b00 = *(const v16h*)(bbase);
    v16h b01 = *(const v16h*)(bbase + 32);
    v16h b10 = *(const v16h*)(bbase + 16 * DIM);
    v16h b11 = *(const v16h*)(bbase + 16 * DIM + 32);

    for (int jj = 0; jj < N_TOTAL / 32; ++jj) {
        const int c0 = jj * 32;

        v8f acc0 = {}; v8f acc1 = {};
        acc0 = __builtin_amdgcn_wmma_f32_16x16x32_f16(false, a0, false, b00,
                                                      (short)0, acc0, false, false);
        acc0 = __builtin_amdgcn_wmma_f32_16x16x32_f16(false, a1, false, b01,
                                                      (short)0, acc0, false, false);
        acc1 = __builtin_amdgcn_wmma_f32_16x16x32_f16(false, a0, false, b10,
                                                      (short)0, acc1, false, false);
        acc1 = __builtin_amdgcn_wmma_f32_16x16x32_f16(false, a1, false, b11,
                                                      (short)0, acc1, false, false);

        // software pipeline: next B tiles stay in flight through the scan
        if (jj + 1 < N_TOTAL / 32) {
            const _Float16* bn = bbase + (size_t)(c0 + 32) * DIM;
            b00 = *(const v16h*)(bn);
            b01 = *(const v16h*)(bn + 32);
            b10 = *(const v16h*)(bn + 16 * DIM);
            b11 = *(const v16h*)(bn + 16 * DIM + 32);
        }

        // C layout: VGPR v -> (M = v + 8*hi, N = m). Transpose through LDS.
#pragma unroll
        for (int v = 0; v < 8; ++v) {
            sim[(v + 8 * hi) * SIMW + m]      = acc0[v];
            sim[(v + 8 * hi) * SIMW + 16 + m] = acc1[v];
        }
        // branch-free self-column poison: non-diag iterations hit pad col 33
        {
            int pc = (jj == diagJJ) ? (grow - c0) : 33;
            sim[m * SIMW + pc] = -1.0e30f;
        }
        wave_lds_fence();

        // lane scans row m, its 16-column half; one max-gate per 16 values
        const float* simRow = &sim[m * SIMW + hi * 16];
        float4 q0 = *(const float4*)(simRow + 0);
        float4 q1 = *(const float4*)(simRow + 4);
        float4 q2 = *(const float4*)(simRow + 8);
        float4 q3 = *(const float4*)(simRow + 12);
        float mx0 = fmaxf(fmaxf(q0.x, q0.y), fmaxf(q0.z, q0.w));
        float mx1 = fmaxf(fmaxf(q1.x, q1.y), fmaxf(q1.z, q1.w));
        float mx2 = fmaxf(fmaxf(q2.x, q2.y), fmaxf(q2.z, q2.w));
        float mx3 = fmaxf(fmaxf(q3.x, q3.y), fmaxf(q3.z, q3.w));
        if (fmaxf(fmaxf(mx0, mx1), fmaxf(mx2, mx3)) > tv[KNB - 1]) {
            int gb = c0 + hi * 16;
            topk_insert(q0.x, gb +  0, tv, ti);
            topk_insert(q0.y, gb +  1, tv, ti);
            topk_insert(q0.z, gb +  2, tv, ti);
            topk_insert(q0.w, gb +  3, tv, ti);
            topk_insert(q1.x, gb +  4, tv, ti);
            topk_insert(q1.y, gb +  5, tv, ti);
            topk_insert(q1.z, gb +  6, tv, ti);
            topk_insert(q1.w, gb +  7, tv, ti);
            topk_insert(q2.x, gb +  8, tv, ti);
            topk_insert(q2.y, gb +  9, tv, ti);
            topk_insert(q2.z, gb + 10, tv, ti);
            topk_insert(q2.w, gb + 11, tv, ti);
            topk_insert(q3.x, gb + 12, tv, ti);
            topk_insert(q3.y, gb + 13, tv, ti);
            topk_insert(q3.z, gb + 14, tv, ti);
            topk_insert(q3.w, gb + 15, tv, ti);
        }
        wave_lds_fence();   // order scan loads before next iter's LDS stores
    }

    // ---- merge the two half-column lists per row, emit edges ----
#pragma unroll
    for (int t = 0; t < KNB; ++t) { mval[lane][t] = tv[t]; midx[lane][t] = ti[t]; }
    __syncthreads();

    if (lane < 16) {
        const int gr = rowBase + lane;
        int p = 0, q = 0;
#pragma unroll
        for (int t = 0; t < KNB; ++t) {
            float va = mval[lane][p], vb = mval[lane + 16][q];
            float v; int id;
            if (va >= vb) { v = va; id = midx[lane][p];      ++p; }
            else          { v = vb; id = midx[lane + 16][q]; ++q; }
            bool valid = id >= 0;                 // only real >THRESH entries
            int e = gr * KNB + t;
            iout[e]            = gr;              // src
            iout[OFF_DST + e]  = valid ? id : -1; // dst
            fout[OFF_W + e]    = valid ? v : 0.0f;
        }
        // self loops
        iout[OFF_SRC_LOOP + gr] = gr;
        iout[OFF_DST_LOOP + gr] = gr;
        fout[OFF_W_LOOP  + gr]  = 1.0f;
    }
}

extern "C" void kernel_launch(void* const* d_in, const int* in_sizes, int n_in,
                              void* d_out, int out_size, void* d_ws, size_t ws_size,
                              hipStream_t stream) {
    const float* x = (const float*)d_in[0];
    _Float16* xh = (_Float16*)d_ws;          // 8192 x 64 f16 = 1 MB scratch
    int*   iout = (int*)d_out;
    float* fout = (float*)d_out;

    normalize_f16_kernel<<<N_TOTAL / 256, 256, 0, stream>>>(x, xh);
    topk_graph_kernel<<<N_TOTAL / 16, 32, 0, stream>>>(xh, iout, fout);
}
